// CombinedLoss_34746285424742
// MI455X (gfx1250) — matrix operand order repaired and checked by reading, not compiled, not run on hardware
//
#include <hip/hip_runtime.h>
#include <hip/hip_bf16.h>
#include <math.h>

typedef __attribute__((ext_vector_type(16))) _Float16 v16h;
typedef __attribute__((ext_vector_type(8)))  float    v8f;

#define BATCH 32
#define IMG_H 512
#define IMG_W 512
#define NPIX  (BATCH * IMG_H * IMG_W)

// ---------------------------------------------------------------- helpers
__device__ __forceinline__ float warp_sum32(float v) {
  #pragma unroll
  for (int m = 16; m >= 1; m >>= 1) v += __shfl_xor(v, m, 32);
  return v;
}

__device__ __forceinline__ float clip01(float x) {
  return fminf(fmaxf(x, 0.f), 1.f);
}

__device__ __forceinline__ float soft_thr(float x, float t) {
  float a = fmaxf(fabsf(x) - t, 0.f);
  return copysignf(a, x);
}

// ---------------------------------------------------------------- L1 kernel
__global__ __launch_bounds__(256) void l1_kernel(const float* __restrict__ p,
                                                 const float* __restrict__ t,
                                                 float* __restrict__ acc, int n) {
  int i = blockIdx.x * blockDim.x + threadIdx.x;
  int stride = gridDim.x * blockDim.x;
  float s = 0.f;
  for (; i < n; i += stride) {
    __builtin_prefetch(p + i + stride, 0, 0);   // global_prefetch_b8
    float pv = clip01(p[i]);
    s += fabsf(pv - t[i]);
  }
  s = warp_sum32(s);
  if ((threadIdx.x & 31) == 0) atomicAdd(&acc[0], s);
}

// ---------------------------------------------------------------- SSIM kernel
// One wave32 per 16x16 output tile. Separable 11-tap Gaussian conv on the WMMA
// pipe: horizontal pass = data(A, 16x32) x Toeplitz(B, 32x16);
// vertical pass = Toeplitz(A, 16x32) x hconv(B, 32x16).
// The two horizontal chunk-WMMAs per field are issued back-to-back so the
// second one covers the first one's WMMA->VALU hazard window (no v_nop stall
// before the f32->f16 conversion chain).
__global__ __launch_bounds__(256) void ssim_kernel(const float* __restrict__ pred,
                                                   const float* __restrict__ targ,
                                                   float* __restrict__ acc) {
  __shared__ _Float16 regP[8][32][32];                     // 16 KB
  __shared__ _Float16 regT[8][32][32];                     // 16 KB
  __shared__ __align__(32) _Float16 hbuf[8][32][16];       //  8 KB

  const int wv   = threadIdx.x >> 5;
  const int lane = threadIdx.x & 31;
  const int tile = blockIdx.x * 8 + wv;                    // 32768 tiles total
  const int tilesPerImg = (IMG_H / 16) * (IMG_W / 16);     // 1024
  const int b  = tile / tilesPerImg;
  const int tr = tile % tilesPerImg;
  const int ty = tr / (IMG_W / 16);
  const int tx = tr % (IMG_W / 16);
  const int y0 = ty * 16 - 5;
  const int x0 = tx * 16 - 5;
  const float* pb = pred + (size_t)b * IMG_H * IMG_W;
  const float* tb = targ + (size_t)b * IMG_H * IMG_W;

  // --- normalized Gaussian coefficient g(d), d in [0,10], sigma=1.5
  float gs = 0.f;
  #pragma unroll
  for (int i = 0; i < 11; i++) { float x = (float)i - 5.f; gs += expf(-x * x * (1.f / 4.5f)); }
  const float ginv = 1.f / gs;

  // --- load 26x26 region (zero padded to 32x32), clip pred, as f16
  #pragma unroll 4
  for (int r = 0; r < 32; r++) {
    int gy = y0 + r, gx = x0 + lane;
    float p = 0.f, t = 0.f;
    if (r < 26 && lane < 26 && gy >= 0 && gy < IMG_H && gx >= 0 && gx < IMG_W) {
      p = clip01(pb[gy * IMG_W + gx]);
      t = tb[gy * IMG_W + gx];
    }
    regP[wv][r][lane] = (_Float16)p;
    regT[wv][r][lane] = (_Float16)t;
  }
  __syncthreads();

  // --- constant filter operands
  // bfilt: B-layout (lane&15 = column n, half h -> K): B[k][n] = g[k-n]
  // afilt: A-layout (lane&15 = row m, half h -> K): A[m][k] = g[k-m]
  v16h bfilt, afilt;
  #pragma unroll
  for (int h = 0; h < 16; h++) {
    int d0 = ((h & 7) + ((h >= 8) ? 16 : 0) + ((lane & 16) ? 8 : 0)) - (lane & 15);
    float xb = (float)d0 - 5.f;
    bfilt[h] = (d0 >= 0 && d0 <= 10) ? (_Float16)(expf(-xb * xb * (1.f / 4.5f)) * ginv)
                                     : (_Float16)0.f;
    int K  = (h & 7) + ((h >= 8) ? 16 : 0) + ((lane & 16) ? 8 : 0);
    int d1 = K - (lane & 15);
    float xa = (float)d1 - 5.f;
    afilt[h] = (d1 >= 0 && d1 <= 10) ? (_Float16)(expf(-xa * xa * (1.f / 4.5f)) * ginv)
                                     : (_Float16)0.f;
  }

  // --- preload this lane's A-operand slots for both 16-row chunks
  _Float16 pv[2][16], tv[2][16];
  #pragma unroll
  for (int ch = 0; ch < 2; ch++) {
    int row = ch * 16 + (lane & 15);
    int c0  = (lane & 16) ? 8 : 0;
    #pragma unroll
    for (int i = 0; i < 8; i++) {
      pv[ch][i]     = regP[wv][row][c0 + i];
      pv[ch][8 + i] = regP[wv][row][c0 + 16 + i];
      tv[ch][i]     = regT[wv][row][c0 + i];
      tv[ch][8 + i] = regT[wv][row][c0 + 16 + i];
    }
  }

  // --- 5 fields: p, t, p*p, t*t, p*t  -> separable conv results
  v8f R[5];
  #pragma unroll
  for (int f = 0; f < 5; f++) {
    // build both chunk A-operands first
    v16h a0, a1;
    #pragma unroll
    for (int i = 0; i < 16; i++) {
      _Float16 p0 = pv[0][i], t0 = tv[0][i];
      _Float16 p1 = pv[1][i], t1 = tv[1][i];
      a0[i] = (f == 0) ? p0
            : (f == 1) ? t0
            : (f == 2) ? (_Float16)(p0 * p0)
            : (f == 3) ? (_Float16)(t0 * t0)
            :            (_Float16)(p0 * t0);
      a1[i] = (f == 0) ? p1
            : (f == 1) ? t1
            : (f == 2) ? (_Float16)(p1 * p1)
            : (f == 3) ? (_Float16)(t1 * t1)
            :            (_Float16)(p1 * t1);
    }
    // issue both horizontal WMMAs back-to-back (independent D regs)
    v8f z0 = {}, z1 = {};
    v8f d0 = __builtin_amdgcn_wmma_f32_16x16x32_f16(false, a0, false, bfilt,
                                                    (short)0, z0, false, false);
    v8f d1 = __builtin_amdgcn_wmma_f32_16x16x32_f16(false, a1, false, bfilt,
                                                    (short)0, z1, false, false);
    // D layout: lane&15 = column, VGPR v = row v (+8 for lanes 16..31)
    int n = lane & 15;
    int rb = (lane & 16) ? 8 : 0;
    #pragma unroll
    for (int v = 0; v < 8; v++) {
      hbuf[wv][rb + v][n]      = (_Float16)d0[v];
      hbuf[wv][16 + rb + v][n] = (_Float16)d1[v];
    }
    __syncthreads();
    v16h bop;   // B-layout: lane&15 = column n, half h -> K
    #pragma unroll
    for (int s = 0; s < 16; s++)
      bop[s] = hbuf[wv][(s & 7) + ((s >= 8) ? 16 : 0) + ((lane & 16) ? 8 : 0)][lane & 15];
    v8f z2 = {};
    R[f] = __builtin_amdgcn_wmma_f32_16x16x32_f16(false, afilt, false, bop,
                                                  (short)0, z2, false, false);
    __syncthreads();
  }

  // --- SSIM map + wave reduction
  const float C1v = 1e-4f, C2v = 9e-4f;
  float ssum = 0.f;
  #pragma unroll
  for (int v = 0; v < 8; v++) {
    float mu1 = R[0][v], mu2 = R[1][v];
    float m11 = mu1 * mu1, m22 = mu2 * mu2, m12 = mu1 * mu2;
    float s1 = R[2][v] - m11, s2 = R[3][v] - m22, s12 = R[4][v] - m12;
    float num = (2.f * m12 + C1v) * (2.f * s12 + C2v);
    float den = (m11 + m22 + C1v) * (s1 + s2 + C2v);
    ssum += num / den;
  }
  ssum = warp_sum32(ssum);
  if (lane == 0) atomicAdd(&acc[1], ssum);
}

// ---------------------------------------------------------------- Wavelet
__device__ __forceinline__ void haar4(float a, float b, float c, float d,
                                      float& cA, float& cH, float& cV, float& cD) {
  cA = (a + b + c + d) * 0.5f;
  cH = (a + b - c - d) * 0.5f;
  cV = (a - b + c - d) * 0.5f;
  cD = (a - b - c + d) * 0.5f;
}

__device__ __forceinline__ float sdiff(float x, float y, float t) {
  return fabsf(soft_thr(x, t) - soft_thr(y, t));
}

// One thread per 8x8 block: the full 3-level Haar pyramid is local.
__global__ __launch_bounds__(256) void wav_kernel(const float* __restrict__ pred,
                                                  const float* __restrict__ targ,
                                                  float* __restrict__ acc) {
  int tid = blockIdx.x * blockDim.x + threadIdx.x;          // 131072 threads
  int b   = tid >> 12;
  int rem = tid & 4095;
  int by  = rem >> 6, bx = rem & 63;
  const float* pb = pred + (size_t)b * IMG_H * IMG_W + (by * 8) * IMG_W + bx * 8;
  const float* tb = targ + (size_t)b * IMG_H * IMG_W + (by * 8) * IMG_W + bx * 8;

  const float thr3 = 50.f / 4.f / 255.f;   // finest details  (level_idx = 3)
  const float thr2 = 50.f / 2.f / 255.f;   // mid             (level_idx = 2)
  const float thr1 = 50.f / 255.f;         // coarsest        (level_idx = 1)

  float cA1p[4][4], cA1t[4][4];
  float s3 = 0.f;
  #pragma unroll
  for (int r = 0; r < 4; r++) {
    float p0[8], p1[8], t0[8], t1[8];
    #pragma unroll
    for (int c = 0; c < 8; c++) {
      p0[c] = clip01(pb[(2 * r) * IMG_W + c]);
      p1[c] = clip01(pb[(2 * r + 1) * IMG_W + c]);
      t0[c] = tb[(2 * r) * IMG_W + c];
      t1[c] = tb[(2 * r + 1) * IMG_W + c];
    }
    #pragma unroll
    for (int c = 0; c < 4; c++) {
      float cAp, cHp, cVp, cDp, cAt, cHt, cVt, cDt;
      haar4(p0[2 * c], p0[2 * c + 1], p1[2 * c], p1[2 * c + 1], cAp, cHp, cVp, cDp);
      haar4(t0[2 * c], t0[2 * c + 1], t1[2 * c], t1[2 * c + 1], cAt, cHt, cVt, cDt);
      s3 += sdiff(cHp, cHt, thr3) + sdiff(cVp, cVt, thr3) + sdiff(cDp, cDt, thr3);
      cA1p[r][c] = cAp; cA1t[r][c] = cAt;
    }
  }
  float cA2p[2][2], cA2t[2][2];
  float s2 = 0.f;
  #pragma unroll
  for (int r = 0; r < 2; r++) {
    #pragma unroll
    for (int c = 0; c < 2; c++) {
      float cAp, cHp, cVp, cDp, cAt, cHt, cVt, cDt;
      haar4(cA1p[2 * r][2 * c], cA1p[2 * r][2 * c + 1],
            cA1p[2 * r + 1][2 * c], cA1p[2 * r + 1][2 * c + 1], cAp, cHp, cVp, cDp);
      haar4(cA1t[2 * r][2 * c], cA1t[2 * r][2 * c + 1],
            cA1t[2 * r + 1][2 * c], cA1t[2 * r + 1][2 * c + 1], cAt, cHt, cVt, cDt);
      s2 += sdiff(cHp, cHt, thr2) + sdiff(cVp, cVt, thr2) + sdiff(cDp, cDt, thr2);
      cA2p[r][c] = cAp; cA2t[r][c] = cAt;
    }
  }
  float s1 = 0.f;
  {
    float cAp, cHp, cVp, cDp, cAt, cHt, cVt, cDt;
    haar4(cA2p[0][0], cA2p[0][1], cA2p[1][0], cA2p[1][1], cAp, cHp, cVp, cDp);
    haar4(cA2t[0][0], cA2t[0][1], cA2t[1][0], cA2t[1][1], cAt, cHt, cVt, cDt);
    s1 += sdiff(cHp, cHt, thr1) + sdiff(cVp, cVt, thr1) + sdiff(cDp, cDt, thr1);
  }
  s1 = warp_sum32(s1);
  s2 = warp_sum32(s2);
  s3 = warp_sum32(s3);
  if ((threadIdx.x & 31) == 0) {
    atomicAdd(&acc[2], s1);
    atomicAdd(&acc[3], s2);
    atomicAdd(&acc[4], s3);
  }
}

// ---------------------------------------------------------------- glue
__global__ void init_kernel(float* acc) {
  if (threadIdx.x < 8) acc[threadIdx.x] = 0.f;
}

__global__ void final_kernel(const float* __restrict__ acc, float* __restrict__ out) {
  if (threadIdx.x != 0 || blockIdx.x != 0) return;
  const float N = (float)NPIX;
  float l1 = acc[0] / N;
  float ssim_mean = acc[1] / N;
  float ssim_loss = fminf(fmaxf(1.f - ssim_mean, 0.f), 2.f);
  // per-detail-array element counts: 32*64^2, 32*128^2, 32*256^2
  float wav = acc[2] / (3.f * 131072.f * 1.f)
            + acc[3] / (3.f * 524288.f * 2.f)
            + acc[4] / (3.f * 2097152.f * 3.f);
  out[0] = 1.0f * l1 + 0.5f * ssim_loss + 0.1f * wav;
}

extern "C" void kernel_launch(void* const* d_in, const int* in_sizes, int n_in,
                              void* d_out, int out_size, void* d_ws, size_t ws_size,
                              hipStream_t stream) {
  (void)in_sizes; (void)n_in; (void)out_size; (void)ws_size;
  const float* pred = (const float*)d_in[0];
  const float* targ = (const float*)d_in[1];
  float* acc = (float*)d_ws;    // acc[0]=l1 acc[1]=ssim acc[2..4]=wavelet levels
  float* out = (float*)d_out;

  init_kernel<<<1, 32, 0, stream>>>(acc);
  l1_kernel<<<1024, 256, 0, stream>>>(pred, targ, acc, NPIX);
  ssim_kernel<<<4096, 256, 0, stream>>>(pred, targ, acc);   // 32768 tiles, 8 waves/block
  wav_kernel<<<512, 256, 0, stream>>>(pred, targ, acc);     // 131072 8x8 blocks
  final_kernel<<<1, 32, 0, stream>>>(acc, out);
}
